// QuantumAttention_72129680769346
// MI455X (gfx1250) — matrix operand order repaired and checked by reading, not compiled, not run on hardware
//
#include <hip/hip_runtime.h>

typedef __attribute__((ext_vector_type(16))) __bf16 v16bf;
typedef __attribute__((ext_vector_type(8)))  float  v8f;
typedef __attribute__((ext_vector_type(4)))  unsigned int v4u;

union FragBF {
    v16bf v;
    v4u   q[2];
};

__device__ __forceinline__ unsigned short f2bf(float f) {
    unsigned int u = __float_as_uint(f);
    u += 0x7FFFu + ((u >> 16) & 1u);   // round-to-nearest-even
    return (unsigned short)(u >> 16);
}

// ---------------------------------------------------------------------------
// Elementwise fp32 -> bf16 convert
// ---------------------------------------------------------------------------
__global__ void cvt_bf16_kernel(const float* __restrict__ in,
                                unsigned short* __restrict__ out, int n) {
    int i = blockIdx.x * blockDim.x + threadIdx.x;
    if (i < n) out[i] = f2bf(in[i]);
}

// ---------------------------------------------------------------------------
// LDS-tiled transpose + convert: VT[d*M + m] = bf16(V[m*D + d])
// ---------------------------------------------------------------------------
__global__ void transpose_cvt_kernel(const float* __restrict__ V,
                                     unsigned short* __restrict__ VT,
                                     int Mrows, int Dcols) {
    __shared__ float tile[32][33];
    const int d0 = blockIdx.x * 32;
    const int m0 = blockIdx.y * 32;
    const int c  = threadIdx.x & 31;
    const int r0 = threadIdx.x >> 5;   // 0..7
    #pragma unroll
    for (int r = r0; r < 32; r += 8)
        tile[r][c] = V[(size_t)(m0 + r) * Dcols + d0 + c];
    __syncthreads();
    #pragma unroll
    for (int r = r0; r < 32; r += 8)
        VT[(size_t)(d0 + r) * Mrows + m0 + c] = f2bf(tile[c][r]);
}

// ---------------------------------------------------------------------------
// Async double-buffered bf16 WMMA GEMM:
//   C[Mtot][Ntot] = A[Mtot][K] * B[Ntot][K]^T   (fp32 out)
// A and B row-major with the contraction dim contiguous.
// 256 threads = 8 waves; wave w computes rows [w*16*RT, w*16*RT + 16*RT).
// Panels staged global->LDS via global_load_async_to_lds_b128 (ASYNCcnt),
// double-buffered so panel k+1 streams in while panel k feeds the WMMAs.
// ---------------------------------------------------------------------------
template <int BM, int BN, int BK>
__global__ void __launch_bounds__(256)
gemm_async_kernel(const unsigned short* __restrict__ A,
                  const unsigned short* __restrict__ B,
                  float* __restrict__ C,
                  int Mtot, int Ntot, int Ktot) {
    static_assert(BK == 32 && BN == 128, "layout assumptions");
    constexpr int RT    = BM / 128;            // 16-row tiles per wave
    constexpr int ACH   = BM * BK * 2 / 16;    // 16B chunks per A panel
    constexpr int BCH   = BN * BK * 2 / 16;
    constexpr int ACPT  = ACH / 256;           // chunks per thread
    constexpr int BCPT  = BCH / 256;
    constexpr int NPEND = ACPT + BCPT;         // async loads issued per panel

    __shared__ __attribute__((aligned(16))) unsigned short sA[2 * BM * BK];
    __shared__ __attribute__((aligned(16))) unsigned short sB[2 * BN * BK];

    const int colBase = blockIdx.x * BN;
    const int rowBase = blockIdx.y * BM;
    const int tid  = threadIdx.x;
    const int wave = tid >> 5;
    const int lane = tid & 31;
    const int laneHi = lane >> 4;
    const int laneLo = lane & 15;

    // Low 32 bits of the generic address of an LDS object = LDS byte offset.
    const unsigned ldsA = (unsigned)(unsigned long long)(&sA[0]);
    const unsigned ldsB = (unsigned)(unsigned long long)(&sB[0]);

    v8f acc[RT][8];
    const v8f vzero = {0.f, 0.f, 0.f, 0.f, 0.f, 0.f, 0.f, 0.f};
    #pragma unroll
    for (int rt = 0; rt < RT; ++rt)
        #pragma unroll
        for (int j = 0; j < 8; ++j) acc[rt][j] = vzero;

    // Stage one k-panel into LDS buffer `buf` with async loads (no VGPR bounce).
    auto stage = [&](int k0, int buf) {
        const unsigned aBase = ldsA + (unsigned)(buf * BM * BK * 2);
        #pragma unroll
        for (int i = 0; i < ACPT; ++i) {
            const int c   = tid + 256 * i;     // BK=32 -> 64B row = 4 chunks
            const int row = c >> 2;
            const int cc  = c & 3;
            unsigned long long ga =
                (unsigned long long)(A + (size_t)(rowBase + row) * Ktot + k0 + cc * 8);
            asm volatile("global_load_async_to_lds_b128 %0, %1, off"
                         :: "v"(aBase + (unsigned)c * 16u), "v"(ga) : "memory");
        }
        const unsigned bBase = ldsB + (unsigned)(buf * BN * BK * 2);
        #pragma unroll
        for (int i = 0; i < BCPT; ++i) {
            const int c   = tid + 256 * i;
            const int row = c >> 2;
            const int cc  = c & 3;
            unsigned long long ga =
                (unsigned long long)(B + (size_t)(colBase + row) * Ktot + k0 + cc * 8);
            asm volatile("global_load_async_to_lds_b128 %0, %1, off"
                         :: "v"(bBase + (unsigned)c * 16u), "v"(ga) : "memory");
        }
    };

    const int P = Ktot / BK;
    stage(0, 0);
    int cur = 0;

    for (int p = 0; p < P; ++p) {
        if (p + 1 < P) {
            stage((p + 1) * BK, cur ^ 1);
            // current panel's NPEND loads done; next panel's still in flight
            asm volatile("s_wait_asynccnt %0" :: "i"(NPEND) : "memory");
        } else {
            asm volatile("s_wait_asynccnt 0" ::: "memory");
        }
        __syncthreads();   // all waves' async loads for this panel visible

        const unsigned short* a_ = sA + (size_t)cur * BM * BK;
        const unsigned short* b_ = sB + (size_t)cur * BN * BK;

        // A fragments: 16x32 bf16, lane = row (laneLo), halves by laneHi
        FragBF fa[RT];
        #pragma unroll
        for (int rt = 0; rt < RT; ++rt) {
            const unsigned short* ar = a_ + (size_t)(wave * (16 * RT) + rt * 16 + laneLo) * BK;
            fa[rt].q[0] = *(const v4u*)(ar + 8 * laneHi);
            fa[rt].q[1] = *(const v4u*)(ar + 16 + 8 * laneHi);
        }
        #pragma unroll
        for (int j = 0; j < 8; ++j) {
            // B fragment: 32x16 bf16, lane = col (laneLo), K = 8*laneHi+0..7, 16+8*laneHi+0..7
            FragBF fb;
            const unsigned short* br = b_ + (size_t)(j * 16 + laneLo) * BK + 8 * laneHi;
            fb.q[0] = *(const v4u*)(br);
            fb.q[1] = *(const v4u*)(br + 16);
            #pragma unroll
            for (int rt = 0; rt < RT; ++rt)   // B fragment reused across row tiles
                acc[rt][j] = __builtin_amdgcn_wmma_f32_16x16x32_bf16(
                    false, fa[rt].v, false, fb.v, (short)0, acc[rt][j], false, false);
        }
        __syncthreads();   // done reading buf `cur` before it is overwritten
        cur ^= 1;
    }

    // C/D layout: VGPR v -> M = v + 8*laneHi, N = laneLo
    #pragma unroll
    for (int rt = 0; rt < RT; ++rt) {
        #pragma unroll
        for (int j = 0; j < 8; ++j) {
            const int col = colBase + j * 16 + laneLo;
            #pragma unroll
            for (int v = 0; v < 8; ++v) {
                const int row = rowBase + wave * (16 * RT) + rt * 16 + 8 * laneHi + v;
                C[(size_t)row * Ntot + col] = acc[rt][j][v];
            }
        }
    }
}

// ---------------------------------------------------------------------------
// Row softmax in place (fp32), plus bf16 copy of P for the second GEMM.
// ---------------------------------------------------------------------------
__global__ void softmax_rows_kernel(float* __restrict__ S,
                                    unsigned short* __restrict__ Pb,
                                    int Mcols) {
    const int row = blockIdx.x;
    float* r = S + (size_t)row * Mcols;
    __shared__ float red[256];
    const int tid = threadIdx.x;

    float mx = -__builtin_inff();
    for (int j = tid; j < Mcols; j += 256) mx = fmaxf(mx, r[j]);
    red[tid] = mx;
    __syncthreads();
    #pragma unroll
    for (int s = 128; s > 0; s >>= 1) {
        if (tid < s) red[tid] = fmaxf(red[tid], red[tid + s]);
        __syncthreads();
    }
    mx = red[0];
    __syncthreads();

    float sum = 0.f;
    for (int j = tid; j < Mcols; j += 256) {
        float e = __expf(r[j] - mx);
        r[j] = e;
        sum += e;
    }
    red[tid] = sum;
    __syncthreads();
    #pragma unroll
    for (int s = 128; s > 0; s >>= 1) {
        if (tid < s) red[tid] += red[tid + s];
        __syncthreads();
    }
    const float inv = 1.0f / red[0];

    unsigned short* pb = Pb + (size_t)row * Mcols;
    for (int j = tid; j < Mcols; j += 256) {
        float p = r[j] * inv;
        r[j] = p;
        pb[j] = f2bf(p);
    }
}

// ---------------------------------------------------------------------------
// Launch
// ---------------------------------------------------------------------------
extern "C" void kernel_launch(void* const* d_in, const int* in_sizes, int n_in,
                              void* d_out, int out_size, void* d_ws, size_t ws_size,
                              hipStream_t stream) {
    (void)in_sizes; (void)n_in; (void)out_size; (void)ws_size;
    constexpr int N = 4096, M = 4096, D = 512;

    const float* Q = (const float*)d_in[0];
    const float* K = (const float*)d_in[1];
    const float* V = (const float*)d_in[2];

    char* ws = (char*)d_ws;
    unsigned short* Qb = (unsigned short*)ws;                                   //  4 MB
    unsigned short* Kb = (unsigned short*)(ws + (size_t)N * D * 2);             //  4 MB
    unsigned short* VT = (unsigned short*)(ws + (size_t)(N + M) * D * 2);       //  4 MB  [D][M]
    unsigned short* Pb = (unsigned short*)(ws + (size_t)(N + 2 * M) * D * 2);   // 32 MB  [N][M]

    float* out = (float*)d_out;                 // [N][D]
    float* S   = out + (size_t)N * D;           // [N][M]  probs (scores scratch)

    cvt_bf16_kernel<<<dim3((N * D) / 256), 256, 0, stream>>>(Q, Qb, N * D);
    cvt_bf16_kernel<<<dim3((M * D) / 256), 256, 0, stream>>>(K, Kb, M * D);
    transpose_cvt_kernel<<<dim3(D / 32, M / 32), 256, 0, stream>>>(V, VT, M, D);

    // S = Q * K^T   (A=[N][D], B=[M][D], C=[N][M]); 512 WGs, 32x128 per wave
    gemm_async_kernel<256, 128, 32>
        <<<dim3(M / 128, N / 256), 256, 0, stream>>>(Qb, Kb, S, N, M, D);

    softmax_rows_kernel<<<dim3(N), 256, 0, stream>>>(S, Pb, M);

    // out = P * V   (A=[N][M], B=VT=[D][M], C=[N][D]); wider grid, 16x128/wave
    gemm_async_kernel<128, 128, 32>
        <<<dim3(D / 128, N / 128), 256, 0, stream>>>(Pb, VT, out, N, D, M);
}
